// GCNLarge_20761871909627
// MI455X (gfx1250) — matrix operand order repaired and checked by reading, not compiled, not run on hardware
//
#include <hip/hip_runtime.h>
#include <hip/hip_bf16.h>

// ---------------------------------------------------------------------------
// GCN (5-layer) for MI455X / gfx1250.
//  - Dense h@W via v_wmma_f32_16x16x32_bf16, fp32->bf16 staged in LDS,
//    register+LDS double-buffered K loop (global loads overlap WMMA).
//  - Edge aggregation via native global_atomic_add_f32 (unsafeAtomicAdd),
//    one wave32 per edge, float4 per lane.
// ---------------------------------------------------------------------------

typedef __attribute__((ext_vector_type(16))) __bf16 v16bf;
typedef __attribute__((ext_vector_type(8)))  __bf16 v8bf;
typedef __attribute__((ext_vector_type(4)))  __bf16 v4bf;
typedef __attribute__((ext_vector_type(8)))  float  v8f;

#define GCN_N     100000
#define GCN_E     1600000
#define GCN_FIN   500
#define GCN_HID   128
#define GCN_NCLS  3

#define LDS_STRIDE 40   // halves per LDS row: 80B => 16B-aligned b128 frag loads,
                        // bank starts 0,20,40,... mod 64 -> conflict-free

// ========================= WMMA GEMM: C[Mx128] = A[MxK] @ W[Kx128] ==========
__global__ __launch_bounds__(256)
void gcn_gemm_bf16_n128(const float* __restrict__ A, const float* __restrict__ W,
                        float* __restrict__ C, int M, int K) {
  __shared__ __attribute__((aligned(16))) __bf16 As[2][128 * LDS_STRIDE];
  __shared__ __attribute__((aligned(16))) __bf16 Bs[2][128 * LDS_STRIDE]; // W^T: [n][k]

  const int t    = threadIdx.x;
  const int lane = t & 31;
  const int wv   = t >> 5;        // wave 0..7 -> rows 16*wv..16*wv+15
  const int hi   = lane >> 4;     // half-wave selector
  const int l15  = lane & 15;
  const int rowBase = blockIdx.x * 128;

  // B staging coords: thread owns column n, k-range [kg*16, kg*16+16)
  const int bn = t & 127;
  const int kg = t >> 7;          // 0 or 1

  float ra[16];                   // A chunk regs: 4 x float4
  float rb[16];                   // B chunk regs: 16 k of one column

  v8f acc[8] = {};                // 8 column blocks of 16

  const int nChunks = (K + 31) >> 5;

  // ---- chunk loaders: global -> registers (no LDS traffic) ----
  auto load_regs = [&](int kc) {
    const int k0 = kc << 5;
    if (k0 + 32 <= K) {
      // fast path: unguarded vector loads; rows clamped (stores masked later)
      #pragma unroll
      for (int i = 0; i < 4; ++i) {
        int e = (t + 256 * i) << 2;              // 0..4095, 4-aligned
        int r = e >> 5, k = e & 31;
        int gr = min(rowBase + r, M - 1);
        const float4 v = *(const float4*)&A[(long long)gr * K + k0 + k];
        ra[4 * i + 0] = v.x; ra[4 * i + 1] = v.y;
        ra[4 * i + 2] = v.z; ra[4 * i + 3] = v.w;
      }
      #pragma unroll
      for (int i = 0; i < 16; ++i)                // coalesced: lanes sweep n
        rb[i] = W[(long long)(k0 + kg * 16 + i) * 128 + bn];
    } else {
      // partial final chunk (layer 1 only): guarded scalar fills
      #pragma unroll
      for (int i = 0; i < 4; ++i) {
        int e = (t + 256 * i) << 2;
        int r = e >> 5, k = e & 31;
        int gr = min(rowBase + r, M - 1);
        #pragma unroll
        for (int c = 0; c < 4; ++c) {
          int gk = k0 + k + c;
          ra[4 * i + c] = (gk < K) ? A[(long long)gr * K + gk] : 0.0f;
        }
      }
      #pragma unroll
      for (int i = 0; i < 16; ++i) {
        int gk = k0 + kg * 16 + i;
        rb[i] = (gk < K) ? W[(long long)gk * 128 + bn] : 0.0f;
      }
    }
  };

  // ---- registers -> LDS (bf16 convert, b64 stores) ----
  auto store_lds = [&](int buf) {
    #pragma unroll
    for (int i = 0; i < 4; ++i) {
      int e = (t + 256 * i) << 2;
      int r = e >> 5, k = e & 31;
      v4bf p;
      #pragma unroll
      for (int c = 0; c < 4; ++c) p[c] = (__bf16)ra[4 * i + c];
      *(v4bf*)&As[buf][r * LDS_STRIDE + k] = p;
    }
    #pragma unroll
    for (int i = 0; i < 4; ++i) {
      v4bf p;
      #pragma unroll
      for (int c = 0; c < 4; ++c) p[c] = (__bf16)rb[4 * i + c];
      *(v4bf*)&Bs[buf][bn * LDS_STRIDE + kg * 16 + 4 * i] = p;
    }
  };

  // ---- prologue ----
  load_regs(0);
  store_lds(0);
  __syncthreads();

  for (int kc = 0; kc < nChunks; ++kc) {
    const int cur = kc & 1;
    const bool hasNext = (kc + 1) < nChunks;
    if (hasNext) load_regs(kc + 1);   // global loads in flight during WMMA

    // ---- A fragment (ISA 16-bit A 16x32 layout):
    // lanes0-15: halves0-7 = K0..7, halves8-15 = K16..23 ; lanes16-31: +8
    const int arow = 16 * wv + l15;
    const int kA0  = hi ? 8 : 0;
    const int kA1  = hi ? 24 : 16;
    v8bf alo = *(const v8bf*)&As[cur][arow * LDS_STRIDE + kA0];
    v8bf ahi = *(const v8bf*)&As[cur][arow * LDS_STRIDE + kA1];
    v16bf afrag;
    #pragma unroll
    for (int i = 0; i < 8; ++i) { afrag[i] = alo[i]; afrag[8 + i] = ahi[i]; }

    // ---- B fragments (B 32x16: lane = N, halves0-7 = K0..7, halves8-15 = K16..23 (+8 for hi lanes))
    const int kB0 = 8 * hi;
    #pragma unroll
    for (int cb = 0; cb < 8; ++cb) {
      const int ncol = 16 * cb + l15;
      v8bf blo = *(const v8bf*)&Bs[cur][ncol * LDS_STRIDE + kB0];
      v8bf bhi = *(const v8bf*)&Bs[cur][ncol * LDS_STRIDE + kB0 + 16];
      v16bf bfrag;
      #pragma unroll
      for (int i = 0; i < 8; ++i) { bfrag[i] = blo[i]; bfrag[8 + i] = bhi[i]; }
      acc[cb] = __builtin_amdgcn_wmma_f32_16x16x32_bf16(
          false, afrag, false, bfrag, (short)0, acc[cb], false, false);
    }

    if (hasNext) store_lds(cur ^ 1);  // fill the other buffer
    __syncthreads();
  }

  // ---- store C: v8f elem r -> row 16*wv + r + 8*hi, col 16*cb + l15 ----
  const int rloc = 16 * wv + (hi ? 8 : 0);
  if (rowBase + 128 <= M) {
    #pragma unroll
    for (int cb = 0; cb < 8; ++cb)
      #pragma unroll
      for (int r = 0; r < 8; ++r)
        C[(long long)(rowBase + rloc + r) * 128 + 16 * cb + l15] = acc[cb][r];
  } else {
    #pragma unroll
    for (int cb = 0; cb < 8; ++cb)
      #pragma unroll
      for (int r = 0; r < 8; ++r) {
        int row = rowBase + rloc + r;
        if (row < M) C[(long long)row * 128 + 16 * cb + l15] = acc[cb][r];
      }
  }
}

// ========================= degree / norm ===================================
__global__ void gcn_dis_init(float* dis, int n) {
  int i = blockIdx.x * 256 + threadIdx.x;
  if (i < n) dis[i] = 1.0f;                 // self-loop contributes 1 to deg
}
__global__ void gcn_deg_scatter(const long long* __restrict__ dst, float* dis, int E) {
  int e = blockIdx.x * 256 + threadIdx.x;
  if (e < E) unsafeAtomicAdd(&dis[(int)dst[e]], 1.0f);
}
__global__ void gcn_deg_rsqrt(float* dis, int n) {
  int i = blockIdx.x * 256 + threadIdx.x;
  if (i < n) dis[i] = rsqrtf(dis[i]);       // deg >= 1 always (self-loops)
}

// ========================= aggregation =====================================
// out[i] = t[i] * dis[i]^2   (self-loop term; also zero-initializes out)
__global__ void gcn_selfloop_init(const float* __restrict__ t,
                                  const float* __restrict__ dis,
                                  float* __restrict__ out, int n, int F) {
  long long i = (long long)blockIdx.x * blockDim.x + threadIdx.x;
  if (i < (long long)n * F) {
    int row = (int)(i / F);
    float d = dis[row];
    out[i] = t[i] * d * d;
  }
}

// one wave per edge, float4 per lane (F=128)
__global__ __launch_bounds__(256)
void gcn_edge_scatter128(const float* __restrict__ t,
                         const long long* __restrict__ src,
                         const long long* __restrict__ dst,
                         const float* __restrict__ dis,
                         float* __restrict__ out, int E) {
  int lane = threadIdx.x & 31;
  int e = blockIdx.x * 8 + (threadIdx.x >> 5);
  if (e >= E) return;
  int s = (int)src[e], d = (int)dst[e];
  float scale = dis[s] * dis[d];
  const float4 v = *(const float4*)(t + (long long)s * 128 + lane * 4);
  float* o = out + (long long)d * 128 + lane * 4;
  unsafeAtomicAdd(o + 0, v.x * scale);
  unsafeAtomicAdd(o + 1, v.y * scale);
  unsafeAtomicAdd(o + 2, v.z * scale);
  unsafeAtomicAdd(o + 3, v.w * scale);
}

__global__ void gcn_bias_act(float* __restrict__ h, const float* __restrict__ b,
                             long long total, int F, int relu) {
  long long i = (long long)blockIdx.x * blockDim.x + threadIdx.x;
  if (i < total) {
    float v = h[i] + b[(int)(i % F)];
    h[i] = relu ? fmaxf(v, 0.0f) : v;
  }
}

// ========================= layer 5 (N = 3) =================================
__global__ __launch_bounds__(256)
void gcn_gemm_n3(const float* __restrict__ H, const float* __restrict__ W5,
                 float* __restrict__ T, int M) {
  __shared__ float Ws[GCN_HID * GCN_NCLS];
  for (int i = threadIdx.x; i < GCN_HID * GCN_NCLS; i += 256) Ws[i] = W5[i];
  __syncthreads();
  int row = blockIdx.x * 256 + threadIdx.x;
  if (row >= M) return;
  const float* h = H + (long long)row * GCN_HID;
  float a0 = 0.f, a1 = 0.f, a2 = 0.f;
  #pragma unroll 8
  for (int k = 0; k < GCN_HID; ++k) {
    float v = h[k];
    a0 += v * Ws[k * 3 + 0];
    a1 += v * Ws[k * 3 + 1];
    a2 += v * Ws[k * 3 + 2];
  }
  T[row * 3 + 0] = a0; T[row * 3 + 1] = a1; T[row * 3 + 2] = a2;
}

__global__ void gcn_edge_scatter3(const float* __restrict__ t,
                                  const long long* __restrict__ src,
                                  const long long* __restrict__ dst,
                                  const float* __restrict__ dis,
                                  float* __restrict__ out, int E) {
  int e = blockIdx.x * 256 + threadIdx.x;
  if (e >= E) return;
  int s = (int)src[e], d = (int)dst[e];
  float scale = dis[s] * dis[d];
  #pragma unroll
  for (int c = 0; c < 3; ++c)
    unsafeAtomicAdd(&out[(long long)d * 3 + c], t[(long long)s * 3 + c] * scale);
}

// ========================= host side =======================================
static inline int ceil_div(long long a, long long b) { return (int)((a + b - 1) / b); }

extern "C" void kernel_launch(void* const* d_in, const int* in_sizes, int n_in,
                              void* d_out, int out_size, void* d_ws, size_t ws_size,
                              hipStream_t stream) {
  (void)n_in; (void)out_size; (void)ws_size;
  const float*     x  = (const float*)d_in[0];
  const long long* ei = (const long long*)d_in[1];
  const int N = in_sizes[0] / GCN_FIN;
  const int E = in_sizes[1] / 2;
  const long long* src = ei;          // edge_index[0]
  const long long* dst = ei + E;      // edge_index[1]

  const float* Wl[5] = { (const float*)d_in[2], (const float*)d_in[4],
                         (const float*)d_in[6], (const float*)d_in[8],
                         (const float*)d_in[10] };
  const float* bl[5] = { (const float*)d_in[3], (const float*)d_in[5],
                         (const float*)d_in[7], (const float*)d_in[9],
                         (const float*)d_in[11] };

  // workspace: dis[N] | bufT[N*128] | bufH[N*128]
  char* ws = (char*)d_ws;
  float* dis  = (float*)ws;
  size_t off  = ((size_t)N * sizeof(float) + 255) & ~(size_t)255;
  float* bufT = (float*)(ws + off);
  off += ((size_t)N * GCN_HID * sizeof(float) + 255) & ~(size_t)255;
  float* bufH = (float*)(ws + off);
  float* out  = (float*)d_out;

  // ---- degree / symmetric norm ----
  gcn_dis_init   <<<ceil_div(N, 256), 256, 0, stream>>>(dis, N);
  gcn_deg_scatter<<<ceil_div(E, 256), 256, 0, stream>>>(dst, dis, E);
  gcn_deg_rsqrt  <<<ceil_div(N, 256), 256, 0, stream>>>(dis, N);

  const int gemmGrid = ceil_div(N, 128);
  const long long nf = (long long)N * GCN_HID;

  // ---- layers 1..4: GEMM(WMMA bf16) -> self-loop init -> edge scatter -> bias+relu
  const float* hin = x;
  int K = GCN_FIN;
  for (int l = 0; l < 4; ++l) {
    gcn_gemm_bf16_n128 <<<gemmGrid, 256, 0, stream>>>(hin, Wl[l], bufT, N, K);
    gcn_selfloop_init  <<<ceil_div(nf, 256), 256, 0, stream>>>(bufT, dis, bufH, N, GCN_HID);
    gcn_edge_scatter128<<<ceil_div(E, 8), 256, 0, stream>>>(bufT, src, dst, dis, bufH, E);
    gcn_bias_act       <<<ceil_div(nf, 256), 256, 0, stream>>>(bufH, bl[l], nf, GCN_HID, 1);
    hin = bufH;
    K = GCN_HID;
  }

  // ---- layer 5 (N=3): small GEMM -> aggregate into d_out -> bias (no relu)
  const long long nf3 = (long long)N * GCN_NCLS;
  gcn_gemm_n3       <<<ceil_div(N, 256), 256, 0, stream>>>(bufH, Wl[4], bufT, N);
  gcn_selfloop_init <<<ceil_div(nf3, 256), 256, 0, stream>>>(bufT, dis, out, N, GCN_NCLS);
  gcn_edge_scatter3 <<<ceil_div(E, 256), 256, 0, stream>>>(bufT, src, dst, dis, out, E);
  gcn_bias_act      <<<ceil_div(nf3, 256), 256, 0, stream>>>(out, bl[4], nf3, GCN_NCLS, 0);
}
